// Net_69183333204554
// MI455X (gfx1250) — compile-verified, matrix operand order checked by reading, not run on hardware
//
#include <hip/hip_runtime.h>
#include <math.h>

// ---------------- problem constants ----------------
#define NN   32768
#define EE   131072
#define BB   8
#define FIN  1025
#define KP1  1056      // FIN padded to multiple of 32
#define CH   512
#define NEGV (-1e30f)
#define EPSB (1e-5f)

typedef unsigned long long ull;

// ---------------- WMMA / vector types ----------------
typedef __attribute__((ext_vector_type(16))) __bf16          v16bf;
typedef __attribute__((ext_vector_type(8)))  float           v8f;
typedef __attribute__((ext_vector_type(8)))  unsigned short  u16x8;
typedef __attribute__((ext_vector_type(4)))  unsigned int    v4u;
typedef __attribute__((ext_vector_type(8)))  int             v8i;
typedef __attribute__((ext_vector_type(4)))  int             v4i;

union FragBF { u16x8 p[2]; v16bf v; };

#if defined(__has_builtin)
#  if __has_builtin(__builtin_amdgcn_tensor_load_to_lds) && __has_builtin(__builtin_amdgcn_s_wait_tensorcnt)
#    define USE_TDM 1
#  else
#    define USE_TDM 0
#  endif
#else
#  define USE_TDM 0
#endif

__device__ __forceinline__ unsigned short f2bf(float f) {
    unsigned int u = __float_as_uint(f);
    unsigned int r = u + 0x7FFFu + ((u >> 16) & 1u);   // round-to-nearest-even
    return (unsigned short)(r >> 16);
}

__device__ __forceinline__ void atomicMaxF(float* addr, float val) {
    int* ia = (int*)addr;
    int old = __float_as_int(*addr);
    while (__int_as_float(old) < val) {
        int assumed = old;
        old = atomicCAS(ia, assumed, __float_as_int(val));
        if (old == assumed) break;
    }
}

// ---------------- small utility kernels ----------------
__global__ void fill_f32(float* p, float v, size_t n) {
    size_t i = (size_t)blockIdx.x * blockDim.x + threadIdx.x;
    if (i < n) p[i] = v;
}
__global__ void fill_i32(int* p, int v, size_t n) {
    size_t i = (size_t)blockIdx.x * blockDim.x + threadIdx.x;
    if (i < n) p[i] = v;
}
__global__ void fill_u64(ull* p, ull v, size_t n) {
    size_t i = (size_t)blockIdx.x * blockDim.x + threadIdx.x;
    if (i < n) p[i] = v;
}

// ---------------- edge scoring ----------------
__global__ void node_dots(const float* __restrict__ x, const float* __restrict__ pw,
                          float* s1, float* s2) {
    __shared__ float r1[256], r2[256];
    int n = blockIdx.x;
    const float* row = x + (size_t)n * FIN;
    float a = 0.f, b = 0.f;
    for (int i = threadIdx.x; i < FIN; i += 256) {
        float xv = row[i];
        a += xv * pw[i];
        b += xv * pw[FIN + i];
    }
    r1[threadIdx.x] = a; r2[threadIdx.x] = b;
    __syncthreads();
    for (int s = 128; s > 0; s >>= 1) {
        if (threadIdx.x < s) { r1[threadIdx.x] += r1[threadIdx.x + s]; r2[threadIdx.x] += r2[threadIdx.x + s]; }
        __syncthreads();
    }
    if (threadIdx.x == 0) { s1[n] = r1[0]; s2[n] = r2[0]; }
}

__global__ void edge_e(const int* u, const int* v, const float* s1, const float* s2,
                       const float* pb, float* e, float* m) {
    int k = blockIdx.x * blockDim.x + threadIdx.x;
    if (k >= EE) return;
    float val = s1[u[k]] + s2[v[k]] + pb[0];
    e[k] = val;
    atomicMaxF(&m[v[k]], val);
}

__global__ void edge_expsum(const int* v, const float* e, const float* m, float* ssum) {
    int k = blockIdx.x * blockDim.x + threadIdx.x;
    if (k >= EE) return;
    atomicAdd(&ssum[v[k]], expf(e[k] - m[v[k]]));
}

__global__ void edge_score_keys(const int* v, const float* e, const float* m,
                                const float* ssum, float* score, ull* keys) {
    int k = blockIdx.x * blockDim.x + threadIdx.x;
    if (k >= EE) return;
    float sc = expf(e[k] - m[v[k]]) / ssum[v[k]] + 0.5f;
    score[k] = sc;
    unsigned int fb = __float_as_uint(sc);
    unsigned int asc = (fb & 0x80000000u) ? ~fb : (fb | 0x80000000u);
    unsigned int dsc = ~asc;
    keys[k] = ((ull)dsc << 32) | (unsigned int)k;
}

// ---------------- bitonic sort ----------------
__global__ void bitonic_step(ull* keys, int j, int k) {
    int i = blockIdx.x * blockDim.x + threadIdx.x;
    if (i >= EE) return;
    int ixj = i ^ j;
    if (ixj > i) {
        bool up = ((i & k) == 0);
        ull a = keys[i], b = keys[ixj];
        if ((a > b) == up) { keys[i] = b; keys[ixj] = a; }
    }
}

// ---------------- greedy maximal matching (sequential) ----------------
__global__ void greedy_match(const ull* keys, const int* u, const int* v,
                             int* used, int* sel) {
    if (blockIdx.x != 0 || threadIdx.x != 0) return;
    for (int t = 0; t < EE; ++t) {
        int k = (int)(keys[t] & 0xFFFFFFFFull);
        int uu = u[k], vv = v[k];
        int ok = (!used[uu] && !used[vv]) ? 1 : 0;
        if (ok) { used[uu] = 1; used[vv] = 1; }
        sel[k] = ok;
    }
}

__global__ void cluster_init(int* cluster, float* nscore) {
    int n = blockIdx.x * blockDim.x + threadIdx.x;
    if (n >= NN) return;
    cluster[n] = n;
    nscore[n] = 1.0f;
}

__global__ void apply_sel(const int* sel, const int* u, const int* v,
                          const float* score, int* cluster, float* nscore) {
    int k = blockIdx.x * blockDim.x + threadIdx.x;
    if (k >= EE) return;
    if (!sel[k]) return;
    int uu = u[k], vv = v[k];
    int rep = uu < vv ? uu : vv;
    cluster[uu] = rep;
    cluster[vv] = rep;
    nscore[rep] = score[k];
}

__global__ void scatter_px(const float* __restrict__ x, const int* cluster, float* px) {
    size_t idx = (size_t)blockIdx.x * blockDim.x + threadIdx.x;
    if (idx >= (size_t)NN * FIN) return;
    int n = (int)(idx / FIN);
    int f = (int)(idx % FIN);
    atomicAdd(&px[(size_t)cluster[n] * FIN + f], x[idx]);
}

__global__ void scale_px(float* px, const float* nscore) {
    size_t idx = (size_t)blockIdx.x * blockDim.x + threadIdx.x;
    if (idx >= (size_t)NN * FIN) return;
    px[idx] *= nscore[idx / FIN];
}

__global__ void mark_valid(const int* cluster, int* valid) {
    int n = blockIdx.x * blockDim.x + threadIdx.x;
    if (n >= NN) return;
    valid[n] = (cluster[n] == n) ? 1 : 0;
}

__global__ void coarse_edges(const int* u, const int* v, const int* cluster,
                             int* cu, int* cv) {
    int k = blockIdx.x * blockDim.x + threadIdx.x;
    if (k >= EE) return;
    cu[k] = cluster[u[k]];
    cv[k] = cluster[v[k]];
}

// hash-set dedup (duplicates contribute identically; one winner per pair is exact)
#define TBL_BITS 19
#define TBL_SIZE (1 << TBL_BITS)
__global__ void dedup_edges(const int* cu, const int* cv, float* wE, ull* table) {
    int k = blockIdx.x * blockDim.x + threadIdx.x;
    if (k >= EE) return;
    int a = cu[k], b = cv[k];
    if (a == b) { wE[k] = 0.f; return; }
    ull key = (ull)a * (ull)NN + (ull)b;
    ull h = (key * 0x9E3779B97F4A7C15ull) >> (64 - TBL_BITS);
    const ull EMPTY = ~0ull;
    for (int p = 0; p < TBL_SIZE; ++p) {
        ull slot = (h + p) & (TBL_SIZE - 1);
        ull prev = atomicCAS(&table[slot], EMPTY, key);
        if (prev == EMPTY) { wE[k] = 1.f; return; }
        if (prev == key)   { wE[k] = 0.f; return; }
    }
    wE[k] = 0.f;
}

__global__ void deg_acc(const int* cu, const float* wE, float* deg) {
    int k = blockIdx.x * blockDim.x + threadIdx.x;
    if (k >= EE) return;
    if (wE[k] != 0.f) atomicAdd(&deg[cu[k]], wE[k]);
}

__global__ void deg_inv(const float* deg, float* dis) {
    int n = blockIdx.x * blockDim.x + threadIdx.x;
    if (n >= NN) return;
    float d = deg[n];
    dis[n] = (d > 0.f) ? rsqrtf(fmaxf(d, 1e-12f)) : 0.f;
}

__global__ void edge_nrm(const int* cu, const int* cv, const float* wE,
                         const float* dis, float* nrm) {
    int k = blockIdx.x * blockDim.x + threadIdx.x;
    if (k >= EE) return;
    nrm[k] = -wE[k] * dis[cu[k]] * dis[cv[k]];
}

__global__ void spmm(const int* cu, const int* cv, const float* nrm,
                     const float* __restrict__ h, float* tx, int F) {
    int k = blockIdx.x;
    float c = nrm[k];
    if (c == 0.f) return;
    const float* hr = h + (size_t)cu[k] * F;
    float* o = tx + (size_t)cv[k] * F;
    for (int f = threadIdx.x; f < F; f += blockDim.x)
        atomicAdd(&o[f], c * hr[f]);
}

// ---------------- bf16 conversions (with K zero-padding) ----------------
__global__ void cvt_mat_bf16(const float* in, unsigned short* out, int M, int K, int KP) {
    size_t idx = (size_t)blockIdx.x * blockDim.x + threadIdx.x;
    if (idx >= (size_t)M * KP) return;
    int m = (int)(idx / KP);
    int k = (int)(idx % KP);
    out[idx] = (k < K) ? f2bf(in[(size_t)m * K + k]) : (unsigned short)0;
}
__global__ void cvt_w_bf16(const float* in, unsigned short* out, int K, int KP, int Ncol) {
    size_t idx = (size_t)blockIdx.x * blockDim.x + threadIdx.x;
    if (idx >= (size_t)KP * Ncol) return;
    int k = (int)(idx / Ncol);
    out[idx] = (k < K) ? f2bf(in[idx]) : (unsigned short)0;
}

// ---------------- TDM descriptor issue: 2D tile (32 K-contig x 128 rows) -> LDS ----
#if USE_TDM
__device__ __forceinline__ void tdm_load_tile_a(const unsigned short* gsrc,
                                                unsigned int lds_off, int Kpad) {
    unsigned long long ga = (unsigned long long)gsrc;
    v4u g0;
    g0[0] = 1u;                                                 // count=1, user mode
    g0[1] = lds_off;                                            // lds_addr (bytes)
    g0[2] = (unsigned int)(ga & 0xFFFFFFFFull);                 // global_addr[31:0]
    g0[3] = (unsigned int)((ga >> 32) & 0x01FFFFFFull)          // global_addr[56:32]
          | (2u << 30);                                         // type=2 (image)
    v8i g1;
    g1[0] = (int)(1u << 16);            // workgroup_mask=0, data_size=1 (2 bytes)
    g1[1] = (int)(32u << 16);           // tensor_dim0 = 32
    g1[2] = (int)(128u << 16);          // tensor_dim1 lo16 = 128
    g1[3] = (int)(32u << 16);           // tile_dim0 = 32
    g1[4] = 128;                        // tile_dim1 = 128, tile_dim2 = 0
    g1[5] = Kpad;                       // tensor_dim0_stride (elements)
    g1[6] = 0;
    g1[7] = 0;
    v4i z = {0, 0, 0, 0};
#if __clang_major__ >= 23
    v8i z8 = {0, 0, 0, 0, 0, 0, 0, 0};
    __builtin_amdgcn_tensor_load_to_lds(g0, g1, z, z, z8, 0);
#else
    __builtin_amdgcn_tensor_load_to_lds(g0, g1, z, z, 0);
#endif
}
#endif

// ---------------- fused dual-A WMMA GEMM: C = A1*W1 + A2*W2 + bias ----------------
// 256 thr = 8 waves; block tile 128x16; wave w owns rows [16w,16w+16) x 16 cols.
// Double-buffered LDS tiles; TDM prefetches tile t+1 while tile t computes.
__global__ void gemm_dual_bf16(const unsigned short* __restrict__ A1,
                               const unsigned short* __restrict__ A2,
                               const unsigned short* __restrict__ B1,
                               const unsigned short* __restrict__ B2,
                               const float* __restrict__ bias,
                               float* __restrict__ C,
                               int M, int Kpad, int Ncol) {
    __shared__ unsigned short sA1[2][128 * 32];   // row-major [row][k]
    __shared__ unsigned short sA2[2][128 * 32];
    __shared__ unsigned short sBT1[2][16 * 32];   // transposed [n][k]
    __shared__ unsigned short sBT2[2][16 * 32];

    const int tid  = threadIdx.x;
    const int wv   = __builtin_amdgcn_readfirstlane(tid >> 5);  // scalar wave id
    const int lane = tid & 31;
    const int bm = blockIdx.x * 128;
    const int bn = blockIdx.y * 16;

    const int mloc = (wv << 4) + (lane & 15);
    const int half = lane >> 4;
    const int ncol = lane & 15;

    const size_t abase = (size_t)bm * Kpad;
    const int nk = Kpad >> 5;

#if USE_TDM
    const unsigned int ldsA1b[2] = {
        (unsigned int)(unsigned long long)(void*)&sA1[0][0],
        (unsigned int)(unsigned long long)(void*)&sA1[1][0] };
    const unsigned int ldsA2b[2] = {
        (unsigned int)(unsigned long long)(void*)&sA2[0][0],
        (unsigned int)(unsigned long long)(void*)&sA2[1][0] };
    // prologue: DMA tile 0 into buffer 0
    if (wv == 0) tdm_load_tile_a(A1 + abase, ldsA1b[0], Kpad);
    if (wv == 1) tdm_load_tile_a(A2 + abase, ldsA2b[0], Kpad);
#endif

    v8f acc = {};

    for (int t = 0; t < nk; ++t) {
        const int buf = t & 1;
        const int k0 = t << 5;
#if !USE_TDM
        // fallback: vectorized manual staging of A tiles for this iteration
        for (int idx = tid; idx < 128 * 4; idx += 256) {
            int r = idx >> 2, cb = (idx & 3) << 3;
            size_t g = abase + (size_t)r * Kpad + (k0 + cb);
            *(u16x8*)&sA1[buf][r * 32 + cb] = *(const u16x8*)&A1[g];
            *(u16x8*)&sA2[buf][r * 32 + cb] = *(const u16x8*)&A2[g];
        }
#endif
        // stage B tiles transposed: sBT[n][k] so fragments are K-contiguous
        for (int idx = tid; idx < 32 * 16; idx += 256) {
            int r = idx >> 4, c = idx & 15;              // r = k row, c = n col
            size_t g = (size_t)(k0 + r) * Ncol + (bn + c);
            sBT1[buf][c * 32 + r] = B1[g];
            sBT2[buf][c * 32 + r] = B2[g];
        }
#if USE_TDM
        if (t + 1 < nk) {
            // prefetch next tile into the other buffer; tensor ops complete
            // in order per wave, so <=1 outstanding means tile t has landed.
            const size_t anext = abase + (size_t)(k0 + 32);
            if (wv == 0) tdm_load_tile_a(A1 + anext, ldsA1b[buf ^ 1], Kpad);
            if (wv == 1) tdm_load_tile_a(A2 + anext, ldsA2b[buf ^ 1], Kpad);
            __builtin_amdgcn_s_wait_tensorcnt(1);
        } else {
            __builtin_amdgcn_s_wait_tensorcnt(0);
        }
#endif
        __syncthreads();

        // CDNA5 16-bit fragments: two contiguous 8-element (16B) runs per lane.
        FragBF a1, a2, b1, b2;
        {
            const unsigned short* ra1 = &sA1[buf][mloc * 32];
            const unsigned short* ra2 = &sA2[buf][mloc * 32];
            const unsigned short* rb1 = &sBT1[buf][ncol * 32];
            const unsigned short* rb2 = &sBT2[buf][ncol * 32];
            const int o0 = half * 8, o1 = 16 + half * 8;
            a1.p[0] = *(const u16x8*)&ra1[o0];  a1.p[1] = *(const u16x8*)&ra1[o1];
            a2.p[0] = *(const u16x8*)&ra2[o0];  a2.p[1] = *(const u16x8*)&ra2[o1];
            b1.p[0] = *(const u16x8*)&rb1[o0];  b1.p[1] = *(const u16x8*)&rb1[o1];
            b2.p[0] = *(const u16x8*)&rb2[o0];  b2.p[1] = *(const u16x8*)&rb2[o1];
        }
        acc = __builtin_amdgcn_wmma_f32_16x16x32_bf16(false, a1.v, false, b1.v,
                                                      (short)0, acc, false, false);
        acc = __builtin_amdgcn_wmma_f32_16x16x32_bf16(false, a2.v, false, b2.v,
                                                      (short)0, acc, false, false);
        __syncthreads();
    }

    // C/D layout: VGPR r -> M = r + 8*half, N = lane%16
    const int colg = bn + ncol;
    const float bv = bias[colg];
#pragma unroll
    for (int r = 0; r < 8; ++r) {
        int row = bm + (wv << 4) + half * 8 + r;
        if (row < M) C[(size_t)row * Ncol + colg] = acc[r] + bv;
    }
}

// ---------------- BN(eval) + ReLU, in place ----------------
__global__ void bn_relu(float* h, const float* bn, int Ctot, size_t total) {
    size_t idx = (size_t)blockIdx.x * blockDim.x + threadIdx.x;
    if (idx >= total) return;
    int c = (int)(idx % Ctot);
    float g = bn[c], b = bn[Ctot + c], m = bn[2 * Ctot + c], v = bn[3 * Ctot + c];
    float val = (h[idx] - m) * rsqrtf(v + EPSB) * g + b;
    h[idx] = fmaxf(val, 0.f);
}

// ---------------- graph pooling ----------------
__global__ void pool_acc(const float* __restrict__ h, const int* valid, const int* batch,
                         float* hmax, float* hsum, float* cnt) {
    size_t idx = (size_t)blockIdx.x * blockDim.x + threadIdx.x;
    if (idx >= (size_t)NN * CH) return;
    int n = (int)(idx / CH);
    int c = (int)(idx % CH);
    int b = batch[n];
    int vld = valid[n];
    float hv = h[idx];
    atomicMaxF(&hmax[b * CH + c], vld ? hv : NEGV);
    if (vld) atomicAdd(&hsum[b * CH + c], hv);
    if (c == 0 && vld) atomicAdd(&cnt[b], 1.f);
}

__global__ void build_g(const float* hmax, const float* hsum, const float* cnt, float* g) {
    int idx = blockIdx.x * blockDim.x + threadIdx.x;
    if (idx >= BB * 1024) return;
    int b = idx >> 10, c = idx & 1023;
    g[idx] = (c < CH) ? hmax[b * CH + c]
                      : hsum[b * CH + (c - CH)] / fmaxf(cnt[b], 1.f);
}

// ---------------- readout MLP ----------------
__global__ void lin_bn_relu(const float* in, const float* w, const float* bias,
                            const float* bn, float* out, int In, int Out) {
    int i = blockIdx.x * blockDim.x + threadIdx.x;
    if (i >= BB * Out) return;
    int b = i / Out, o = i % Out;
    const float* r = in + (size_t)b * In;
    float acc = bias[o];
    for (int k = 0; k < In; ++k) acc += r[k] * w[(size_t)k * Out + o];
    float g = bn[o], be = bn[Out + o], m = bn[2 * Out + o], v = bn[3 * Out + o];
    acc = (acc - m) * rsqrtf(v + EPSB) * g + be;
    out[i] = fmaxf(acc, 0.f);
}

__global__ void head_logsoftmax(const float* feat, const float* w, const float* bias,
                                float* out) {
    int b = threadIdx.x;
    if (b >= BB) return;
    float o[4];
#pragma unroll
    for (int j = 0; j < 4; ++j) {
        float a = bias[j];
        for (int k = 0; k < 256; ++k) a += feat[b * 256 + k] * w[k * 4 + j];
        o[j] = fmaxf(a, 0.f);
    }
    float mx = fmaxf(fmaxf(o[0], o[1]), fmaxf(o[2], o[3]));
    float s = 0.f;
#pragma unroll
    for (int j = 0; j < 4; ++j) s += expf(o[j] - mx);
    float ls = logf(s) + mx;
#pragma unroll
    for (int j = 0; j < 4; ++j) out[b * 4 + j] = o[j] - ls;
}

// ---------------- host orchestration ----------------
static inline unsigned int blocks(size_t n, int t) { return (unsigned int)((n + t - 1) / t); }

extern "C" void kernel_launch(void* const* d_in, const int* in_sizes, int n_in,
                              void* d_out, int out_size, void* d_ws, size_t ws_size,
                              hipStream_t stream) {
    (void)in_sizes; (void)n_in; (void)out_size; (void)ws_size;

    const float* x      = (const float*)d_in[0];
    const int*   ei     = (const int*)  d_in[1];
    const int*   batch  = (const int*)  d_in[2];
    const float* pool_w = (const float*)d_in[3];
    const float* pool_b = (const float*)d_in[4];
    const float* c1_w0  = (const float*)d_in[5];
    const float* c1_w1  = (const float*)d_in[6];
    const float* c1_b   = (const float*)d_in[7];
    const float* c2_w0  = (const float*)d_in[8];
    const float* c2_w1  = (const float*)d_in[9];
    const float* c2_b   = (const float*)d_in[10];
    const float* bn1    = (const float*)d_in[11];
    const float* bn2    = (const float*)d_in[12];
    const float* bn3    = (const float*)d_in[13];
    const float* bn4    = (const float*)d_in[14];
    const float* lin1_w = (const float*)d_in[15];
    const float* lin1_b = (const float*)d_in[16];
    const float* lin2_w = (const float*)d_in[17];
    const float* lin2_b = (const float*)d_in[18];
    const float* lin3_w = (const float*)d_in[19];
    const float* lin3_b = (const float*)d_in[20];

    const int* u = ei;
    const int* v = ei + EE;
    float* outp = (float*)d_out;
    float* feat = outp + BB * 4;

    char*  wsb = (char*)d_ws;
    size_t off = 0;
    auto alloc = [&](size_t bytes) -> char* {
        off = (off + 255) & ~(size_t)255;
        char* p = wsb + off;
        off += bytes;
        return p;
    };
    float* s1    = (float*)alloc(NN * 4);
    float* s2    = (float*)alloc(NN * 4);
    float* eArr  = (float*)alloc(EE * 4);
    float* mArr  = (float*)alloc(NN * 4);
    float* ssum  = (float*)alloc(NN * 4);
    float* score = (float*)alloc(EE * 4);
    ull*   keys  = (ull*)  alloc(EE * 8);
    int*   used  = (int*)  alloc(NN * 4);
    int*   sel   = (int*)  alloc(EE * 4);
    int*   clus  = (int*)  alloc(NN * 4);
    float* nsc   = (float*)alloc(NN * 4);
    int*   valid = (int*)  alloc(NN * 4);
    int*   cu    = (int*)  alloc(EE * 4);
    int*   cv    = (int*)  alloc(EE * 4);
    float* wE    = (float*)alloc(EE * 4);
    ull*   table = (ull*)  alloc((size_t)TBL_SIZE * 8);
    float* deg   = (float*)alloc(NN * 4);
    float* dis   = (float*)alloc(NN * 4);
    float* nrm   = (float*)alloc(EE * 4);
    float* px    = (float*)alloc((size_t)NN * FIN * 4);   // reused as h2
    float* tx    = (float*)alloc((size_t)NN * FIN * 4);
    unsigned short* Abf  = (unsigned short*)alloc((size_t)NN * KP1 * 2);
    unsigned short* Tbf  = (unsigned short*)alloc((size_t)NN * KP1 * 2);
    unsigned short* W0bf = (unsigned short*)alloc((size_t)KP1 * CH * 2);
    unsigned short* W1bf = (unsigned short*)alloc((size_t)KP1 * CH * 2);
    float* h1    = (float*)alloc((size_t)NN * CH * 4);
    float* hmax  = (float*)alloc((size_t)BB * CH * 4);
    float* hsum  = (float*)alloc((size_t)BB * CH * 4);
    float* cnt   = (float*)alloc(BB * 4);
    float* gcat  = (float*)alloc((size_t)BB * 1024 * 4);
    float* g1    = (float*)alloc((size_t)BB * CH * 4);

    const int T = 256;

    // ---- edge scores + segment softmax ----
    node_dots<<<NN, T, 0, stream>>>(x, pool_w, s1, s2);
    fill_f32<<<blocks(NN, T), T, 0, stream>>>(mArr, -3.4e38f, NN);
    fill_f32<<<blocks(NN, T), T, 0, stream>>>(ssum, 0.f, NN);
    edge_e<<<blocks(EE, T), T, 0, stream>>>(u, v, s1, s2, pool_b, eArr, mArr);
    edge_expsum<<<blocks(EE, T), T, 0, stream>>>(v, eArr, mArr, ssum);
    edge_score_keys<<<blocks(EE, T), T, 0, stream>>>(v, eArr, mArr, ssum, score, keys);

    // ---- bitonic argsort (descending score, stable) ----
    for (int k = 2; k <= EE; k <<= 1)
        for (int j = k >> 1; j > 0; j >>= 1)
            bitonic_step<<<blocks(EE, T), T, 0, stream>>>(keys, j, k);

    // ---- greedy matching + clustering ----
    fill_i32<<<blocks(NN, T), T, 0, stream>>>(used, 0, NN);
    greedy_match<<<1, 32, 0, stream>>>(keys, u, v, used, sel);
    cluster_init<<<blocks(NN, T), T, 0, stream>>>(clus, nsc);
    apply_sel<<<blocks(EE, T), T, 0, stream>>>(sel, u, v, score, clus, nsc);
    mark_valid<<<blocks(NN, T), T, 0, stream>>>(clus, valid);

    // ---- pooled node features ----
    fill_f32<<<blocks((size_t)NN * FIN, T), T, 0, stream>>>(px, 0.f, (size_t)NN * FIN);
    scatter_px<<<blocks((size_t)NN * FIN, T), T, 0, stream>>>(x, clus, px);
    scale_px<<<blocks((size_t)NN * FIN, T), T, 0, stream>>>(px, nsc);

    // ---- coarsened edges, dedup, normalization ----
    coarse_edges<<<blocks(EE, T), T, 0, stream>>>(u, v, clus, cu, cv);
    fill_u64<<<blocks(TBL_SIZE, T), T, 0, stream>>>(table, ~0ull, TBL_SIZE);
    dedup_edges<<<blocks(EE, T), T, 0, stream>>>(cu, cv, wE, table);
    fill_f32<<<blocks(NN, T), T, 0, stream>>>(deg, 0.f, NN);
    deg_acc<<<blocks(EE, T), T, 0, stream>>>(cu, wE, deg);
    deg_inv<<<blocks(NN, T), T, 0, stream>>>(deg, dis);
    edge_nrm<<<blocks(EE, T), T, 0, stream>>>(cu, cv, wE, dis, nrm);

    // ---- ChebConv layer 1 ----
    fill_f32<<<blocks((size_t)NN * FIN, T), T, 0, stream>>>(tx, 0.f, (size_t)NN * FIN);
    spmm<<<EE, T, 0, stream>>>(cu, cv, nrm, px, tx, FIN);
    cvt_mat_bf16<<<blocks((size_t)NN * KP1, T), T, 0, stream>>>(px, Abf, NN, FIN, KP1);
    cvt_mat_bf16<<<blocks((size_t)NN * KP1, T), T, 0, stream>>>(tx, Tbf, NN, FIN, KP1);
    cvt_w_bf16<<<blocks((size_t)KP1 * CH, T), T, 0, stream>>>(c1_w0, W0bf, FIN, KP1, CH);
    cvt_w_bf16<<<blocks((size_t)KP1 * CH, T), T, 0, stream>>>(c1_w1, W1bf, FIN, KP1, CH);
    {
        dim3 grid(NN / 128, CH / 16);
        gemm_dual_bf16<<<grid, 256, 0, stream>>>(Abf, Tbf, W0bf, W1bf, c1_b, h1, NN, KP1, CH);
    }
    bn_relu<<<blocks((size_t)NN * CH, T), T, 0, stream>>>(h1, bn1, CH, (size_t)NN * CH);

    // ---- ChebConv layer 2 (h2 lives in px) ----
    float* h2 = px;
    fill_f32<<<blocks((size_t)NN * CH, T), T, 0, stream>>>(tx, 0.f, (size_t)NN * CH);
    spmm<<<EE, T, 0, stream>>>(cu, cv, nrm, h1, tx, CH);
    cvt_mat_bf16<<<blocks((size_t)NN * CH, T), T, 0, stream>>>(h1, Abf, NN, CH, CH);
    cvt_mat_bf16<<<blocks((size_t)NN * CH, T), T, 0, stream>>>(tx, Tbf, NN, CH, CH);
    cvt_w_bf16<<<blocks((size_t)CH * CH, T), T, 0, stream>>>(c2_w0, W0bf, CH, CH, CH);
    cvt_w_bf16<<<blocks((size_t)CH * CH, T), T, 0, stream>>>(c2_w1, W1bf, CH, CH, CH);
    {
        dim3 grid(NN / 128, CH / 16);
        gemm_dual_bf16<<<grid, 256, 0, stream>>>(Abf, Tbf, W0bf, W1bf, c2_b, h2, NN, CH, CH);
    }
    bn_relu<<<blocks((size_t)NN * CH, T), T, 0, stream>>>(h2, bn2, CH, (size_t)NN * CH);

    // ---- graph pooling ----
    fill_f32<<<blocks((size_t)BB * CH, T), T, 0, stream>>>(hmax, -3.4e38f, (size_t)BB * CH);
    fill_f32<<<blocks((size_t)BB * CH, T), T, 0, stream>>>(hsum, 0.f, (size_t)BB * CH);
    fill_f32<<<1, T, 0, stream>>>(cnt, 0.f, BB);
    pool_acc<<<blocks((size_t)NN * CH, T), T, 0, stream>>>(h2, valid, batch, hmax, hsum, cnt);
    build_g<<<blocks(BB * 1024, T), T, 0, stream>>>(hmax, hsum, cnt, gcat);

    // ---- readout MLP ----
    lin_bn_relu<<<blocks(BB * 512, T), T, 0, stream>>>(gcat, lin1_w, lin1_b, bn3, g1, 1024, 512);
    lin_bn_relu<<<blocks(BB * 256, T), T, 0, stream>>>(g1, lin2_w, lin2_b, bn4, feat, 512, 256);
    head_logsoftmax<<<1, 32, 0, stream>>>(feat, lin3_w, lin3_b, outp);
}
